// KV_CachedFlashMHADecoderBlock_90615220011031
// MI455X (gfx1250) — matrix operand order repaired and checked by reading, not compiled, not run on hardware
//
#include <hip/hip_runtime.h>
#include <hip/hip_bf16.h>

// ---------------------------------------------------------------------------
// Problem constants (from reference)
// ---------------------------------------------------------------------------
#define BB   2
#define LXD  2048
#define LYD  2048
#define ED   1024
#define HH   16
#define HDD  64
#define FFD  4096

typedef __attribute__((ext_vector_type(16))) __bf16       v16bf;
typedef __attribute__((ext_vector_type(8)))  float        v8f;
typedef __attribute__((ext_vector_type(8)))  unsigned short ushort8;

union FragBF { v16bf v; unsigned int u[8]; };

__device__ __forceinline__ unsigned short f32_to_bf16_rn(float f) {
  union { float f; unsigned int u; } x;
  x.f = f;
  unsigned int r = x.u + 0x7fffu + ((x.u >> 16) & 1u);
  return (unsigned short)(r >> 16);
}

// ---------------------------------------------------------------------------
// fp32 -> bf16 conversion (weights / y)
// ---------------------------------------------------------------------------
__global__ __launch_bounds__(256) void conv_f32_bf16(const float* __restrict__ src,
                                                     unsigned short* __restrict__ dst,
                                                     int n) {
  for (int i = blockIdx.x * 256 + threadIdx.x; i < n; i += gridDim.x * 256)
    dst[i] = f32_to_bf16_rn(src[i]);
}

// ---------------------------------------------------------------------------
// LayerNorm: fp32 row -> bf16 row (gamma/beta fp32)
// ---------------------------------------------------------------------------
__global__ __launch_bounds__(256) void layernorm_bf16(const float* __restrict__ x,
                                                      const float* __restrict__ g,
                                                      const float* __restrict__ b,
                                                      unsigned short* __restrict__ out,
                                                      int Ecols) {
  int row = blockIdx.x;
  const float* xr = x + (size_t)row * Ecols;
  __shared__ float redS[8], redQ[8];
  float s = 0.f, q = 0.f;
  for (int i = threadIdx.x; i < Ecols; i += 256) { float v = xr[i]; s += v; q += v * v; }
#pragma unroll
  for (int m = 1; m < 32; m <<= 1) { s += __shfl_xor(s, m, 32); q += __shfl_xor(q, m, 32); }
  if ((threadIdx.x & 31) == 0) { redS[threadIdx.x >> 5] = s; redQ[threadIdx.x >> 5] = q; }
  __syncthreads();
  s = 0.f; q = 0.f;
#pragma unroll
  for (int w = 0; w < 8; ++w) { s += redS[w]; q += redQ[w]; }
  float mean = s / Ecols;
  float var  = q / Ecols - mean * mean;
  float inv  = rsqrtf(var + 1e-5f);
  for (int i = threadIdx.x; i < Ecols; i += 256)
    out[(size_t)row * Ecols + i] = f32_to_bf16_rn((xr[i] - mean) * inv * g[i] + b[i]);
}

// ---------------------------------------------------------------------------
// RoPE (+ optional 1/sqrt(HD) scale), fp32 in -> bf16 out
// layout: rows x E, col = h*64 + d ; pos = row % L
// ---------------------------------------------------------------------------
__global__ __launch_bounds__(256) void rope_scale_bf16(const float* __restrict__ in,
                                                       unsigned short* __restrict__ out,
                                                       int rows, int L, float scale) {
  int idx = blockIdx.x * 256 + threadIdx.x;          // over rows*H*32 pairs
  int total = rows * HH * 32;
  if (idx >= total) return;
  int j   = idx & 31;
  int h   = (idx >> 5) & 15;
  int row = idx >> 9;
  int pos = row % L;
  float inv = __powf(10000.f, -(float)(2 * j) * (1.f / 64.f));
  float f   = (float)pos * inv;
  float c = __cosf(f), sn = __sinf(f);
  size_t base = (size_t)row * ED + h * 64;
  float x1 = in[base + j], x2 = in[base + 32 + j];
  out[base + j]      = f32_to_bf16_rn((x1 * c - x2 * sn) * scale);
  out[base + 32 + j] = f32_to_bf16_rn((x2 * c + x1 * sn) * scale);
}

// ---------------------------------------------------------------------------
// Tiled WMMA GEMM: C(MxN,f32 acc) = A(MxK,bf16) * B(KxN,bf16) + bias [+res][gelu]
// block = 256 threads = 8 waves (2x4), tile 64x128, K-step 32
// EPI: 0=f32 bias, 1=f32 bias+residual, 2=bf16 bias, 3=bf16 bias+gelu
// ---------------------------------------------------------------------------
template <int EPI>
__global__ __launch_bounds__(256) void gemm_bf16_wmma(const unsigned short* __restrict__ A,
                                                      const unsigned short* __restrict__ Bm,
                                                      const float* __restrict__ bias,
                                                      const float* __restrict__ res,
                                                      void* __restrict__ outp,
                                                      int M, int N, int K) {
  __shared__ __align__(16) unsigned short As[64][40];    // 64 x (32+8)
  __shared__ __align__(16) unsigned short Bs[128][40];   // transposed: [n][k]
  const int tid  = threadIdx.x;
  const int lane = tid & 31, wave = tid >> 5;
  const int wm = wave >> 2, wn = wave & 3;
  const int half = lane >> 4, l16 = lane & 15;
  const int tileM = blockIdx.y * 64;
  const int tileN = blockIdx.x * 128;

  v8f acc[2][2] = {};

  const int ar = tid >> 2;            // A stage row 0..63
  const int ac = (tid & 3) * 8;       // A stage col group

  for (int k0 = 0; k0 < K; k0 += 32) {
    // stage A tile 64x32 (one ushort8 per thread)
    *(ushort8*)&As[ar][ac] =
        *(const ushort8*)(A + (size_t)(tileM + ar) * K + k0 + ac);
    // stage B tile 32x128, written transposed into Bs[n][k]
#pragma unroll
    for (int it = 0; it < 2; ++it) {
      int idx = tid + it * 256;       // 512 chunks of 8
      int kr  = idx >> 4;             // 0..31
      int nc  = (idx & 15) * 8;       // 0..120
      ushort8 vv = *(const ushort8*)(Bm + (size_t)(k0 + kr) * N + tileN + nc);
#pragma unroll
      for (int e = 0; e < 8; ++e) Bs[nc + e][kr] = vv[e];
    }
    if (k0 + 32 < K) {                // prefetch next K tile (global_prefetch_b8)
      __builtin_prefetch(A + (size_t)(tileM + ar) * K + k0 + 32 + ac, 0, 0);
      __builtin_prefetch(Bm + (size_t)(k0 + 32 + (tid >> 4)) * N + tileN + (tid & 15) * 8, 0, 0);
    }
    __syncthreads();

    FragBF af[2], bf[2];
#pragma unroll
    for (int mi = 0; mi < 2; ++mi) {
      int row = wm * 32 + mi * 16 + l16;
#pragma unroll
      for (int i = 0; i < 8; ++i) {
        int kb = (i >> 2) * 16 + half * 8 + (i & 3) * 2;   // 16-bit A 16x32 layout
        af[mi].u[i] = *(const unsigned int*)&As[row][kb];
      }
    }
#pragma unroll
    for (int ni = 0; ni < 2; ++ni) {
      int col = wn * 32 + ni * 16 + l16;
#pragma unroll
      for (int i = 0; i < 8; ++i) {
        int kb = (i >> 2) * 16 + half * 8 + (i & 3) * 2;   // 16-bit B 32x16 layout
        bf[ni].u[i] = *(const unsigned int*)&Bs[col][kb];
      }
    }
#pragma unroll
    for (int mi = 0; mi < 2; ++mi)
#pragma unroll
      for (int ni = 0; ni < 2; ++ni)
        acc[mi][ni] = __builtin_amdgcn_wmma_f32_16x16x32_bf16(
            false, af[mi].v, false, bf[ni].v, (short)0, acc[mi][ni], false, false);
    __syncthreads();
  }

  // epilogue (f32 C/D layout: VGPR r -> row half*8+r, col = lane%16)
#pragma unroll
  for (int mi = 0; mi < 2; ++mi)
#pragma unroll
    for (int ni = 0; ni < 2; ++ni)
#pragma unroll
      for (int r = 0; r < 8; ++r) {
        int row = tileM + wm * 32 + mi * 16 + half * 8 + r;
        int col = tileN + wn * 32 + ni * 16 + l16;
        float v = acc[mi][ni][r] + bias[col];
        size_t o = (size_t)row * N + col;
        if (EPI == 0) {
          ((float*)outp)[o] = v;
        } else if (EPI == 1) {
          ((float*)outp)[o] = v + res[o];
        } else if (EPI == 2) {
          ((unsigned short*)outp)[o] = f32_to_bf16_rn(v);
        } else {  // gelu (tanh approx, matches jax.nn.gelu default)
          float g = 0.5f * v * (1.f + tanhf(0.7978845608f * (v + 0.044715f * v * v * v)));
          ((unsigned short*)outp)[o] = f32_to_bf16_rn(g);
        }
      }
}

// ---------------------------------------------------------------------------
// Flash attention (online softmax), WMMA for S=Q*K^T and O+=P*V.
// Q/K/V: (B, L, H, 64) bf16 (Q pre-scaled & rope'd). Out: (B, Lq, E) bf16.
// block = 128 threads (4 waves); block tile = 64 Q rows; KV tile = 32.
// ---------------------------------------------------------------------------
__global__ __launch_bounds__(128) void attn_wmma(const unsigned short* __restrict__ Qb,
                                                 const unsigned short* __restrict__ Kb,
                                                 const unsigned short* __restrict__ Vb,
                                                 unsigned short* __restrict__ Ob,
                                                 int Lq, int Lk, int causal) {
  // LDS: phase0 Q stage 64x72 (4608); loop: Ks 32x72 (2304) + Vt 64x40 (2560)
  //      + per-wave P scratch 4*16*40 (2560) = 7424 ushorts
  __shared__ __align__(16) unsigned short smem[7424];
  const int b = blockIdx.z, h = blockIdx.y;
  const int qbase = blockIdx.x * 64;
  const int tid = threadIdx.x, lane = tid & 31, wave = tid >> 5;
  const int half = lane >> 4, l16 = lane & 15;

  // ---- stage Q tile (64x64 bf16), then grab loop-invariant A-fragments ----
  unsigned short* Qs = smem;                     // stride 72
#pragma unroll
  for (int it = 0; it < 4; ++it) {
    int idx = tid + it * 128;                    // 512 chunks of 8
    int r = idx >> 3, c = (idx & 7) * 8;
    *(ushort8*)&Qs[r * 72 + c] =
        *(const ushort8*)(Qb + ((((size_t)b * Lq + qbase + r) * HH + h) * HDD) + c);
  }
  __syncthreads();
  FragBF qf[2];
#pragma unroll
  for (int t = 0; t < 2; ++t)
#pragma unroll
    for (int i = 0; i < 8; ++i) {
      int kb = t * 32 + (i >> 2) * 16 + half * 8 + (i & 3) * 2;
      qf[t].u[i] = *(const unsigned int*)&Qs[(wave * 16 + l16) * 72 + kb];
    }
  __syncthreads();

  unsigned short* Ks = smem;                      // 32 x stride 72
  unsigned short* Vt = smem + 32 * 72;            // transposed V: 64(hd) x stride 40
  unsigned short* Ps = smem + 32 * 72 + 64 * 40 + wave * 16 * 40;  // per-wave 16x40

  v8f o[4] = {};
  float mrow[8], lrow[8];
#pragma unroll
  for (int r = 0; r < 8; ++r) { mrow[r] = -1e30f; lrow[r] = 0.f; }

  const int kvTiles = causal ? (qbase / 32 + 2) : (Lk / 32);

  for (int t = 0; t < kvTiles; ++t) {
    const int kv0 = t * 32;
    // ---- stage K (row major, stride 72) and V^T (stride 40) ----
#pragma unroll
    for (int it = 0; it < 2; ++it) {
      int idx = tid + it * 128;                  // 256 chunks of 8
      int r = idx >> 3, c = (idx & 7) * 8;
      *(ushort8*)&Ks[r * 72 + c] =
          *(const ushort8*)(Kb + ((((size_t)b * Lk + kv0 + r) * HH + h) * HDD) + c);
      ushort8 vv = *(const ushort8*)(Vb + ((((size_t)b * Lk + kv0 + r) * HH + h) * HDD) + c);
#pragma unroll
      for (int e = 0; e < 8; ++e) Vt[(c + e) * 40 + r] = vv[e];
    }
    __syncthreads();

    // ---- S = Q * K^T : 16 rows x 32 kv, two N-blocks, two K(HD) chunks ----
    v8f s[2] = {};
#pragma unroll
    for (int nb = 0; nb < 2; ++nb)
#pragma unroll
      for (int c = 0; c < 2; ++c) {
        FragBF kf;
#pragma unroll
        for (int i = 0; i < 8; ++i)
          kf.u[i] = *(const unsigned int*)&Ks[(nb * 16 + l16) * 72 + c * 32 + (i >> 2) * 16 + half * 8 + (i & 3) * 2];
        s[nb] = __builtin_amdgcn_wmma_f32_16x16x32_bf16(
            false, qf[c].v, false, kf.v, (short)0, s[nb], false, false);
      }

    // ---- online softmax update (rows grouped per 16-lane half) ----
#pragma unroll
    for (int r = 0; r < 8; ++r) {
      int qpos = qbase + wave * 16 + half * 8 + r;
      float s0 = s[0][r], s1 = s[1][r];
      if (causal) {
        if (kv0 + l16 > qpos)      s0 = -1e30f;
        if (kv0 + 16 + l16 > qpos) s1 = -1e30f;
      }
      float mx = fmaxf(s0, s1);
#pragma unroll
      for (int msk = 1; msk < 16; msk <<= 1) mx = fmaxf(mx, __shfl_xor(mx, msk, 16));
      float mnew = fmaxf(mrow[r], mx);
      float p0 = __expf(s0 - mnew);
      float p1 = __expf(s1 - mnew);
      float rs = p0 + p1;
#pragma unroll
      for (int msk = 1; msk < 16; msk <<= 1) rs += __shfl_xor(rs, msk, 16);
      float corr = __expf(mrow[r] - mnew);
      lrow[r] = lrow[r] * corr + rs;
      mrow[r] = mnew;
#pragma unroll
      for (int j = 0; j < 4; ++j) o[j][r] *= corr;
      int prow = half * 8 + r;
      Ps[prow * 40 + l16]      = f32_to_bf16_rn(p0);
      Ps[prow * 40 + 16 + l16] = f32_to_bf16_rn(p1);
    }
    // per-wave LDS scratch: ensure P stores land before fragment reads
    asm volatile("s_wait_dscnt 0" ::: "memory");

    // ---- O += P * V  (A = P 16x32, B = V 32x64 via V^T) ----
    FragBF pf;
#pragma unroll
    for (int i = 0; i < 8; ++i) {
      int kb = (i >> 2) * 16 + half * 8 + (i & 3) * 2;
      pf.u[i] = *(const unsigned int*)&Ps[l16 * 40 + kb];
    }
#pragma unroll
    for (int j = 0; j < 4; ++j) {
      FragBF vf;
#pragma unroll
      for (int i = 0; i < 8; ++i)
        vf.u[i] = *(const unsigned int*)&Vt[(j * 16 + l16) * 40 + (i >> 2) * 16 + half * 8 + (i & 3) * 2];
      o[j] = __builtin_amdgcn_wmma_f32_16x16x32_bf16(
          false, pf.v, false, vf.v, (short)0, o[j], false, false);
    }
    __syncthreads();
  }

  // ---- normalize and store bf16 into (B, Lq, E) ----
#pragma unroll
  for (int j = 0; j < 4; ++j)
#pragma unroll
    for (int r = 0; r < 8; ++r) {
      int row = qbase + wave * 16 + half * 8 + r;
      int col = h * 64 + j * 16 + l16;
      Ob[((size_t)b * Lq + row) * ED + col] = f32_to_bf16_rn(o[j][r] / lrow[r]);
    }
}

// ---------------------------------------------------------------------------
// Host orchestration
// ---------------------------------------------------------------------------
extern "C" void kernel_launch(void* const* d_in, const int* in_sizes, int n_in,
                              void* d_out, int out_size, void* d_ws, size_t ws_size,
                              hipStream_t stream) {
  (void)in_sizes; (void)n_in; (void)out_size; (void)ws_size;
  const float* x     = (const float*)d_in[0];
  const float* y     = (const float*)d_in[1];
  const float* ln1_g = (const float*)d_in[2];
  const float* ln1_b = (const float*)d_in[3];
  const float* sa_wq = (const float*)d_in[4];
  const float* sa_bq = (const float*)d_in[5];
  const float* sa_wk = (const float*)d_in[6];
  const float* sa_bk = (const float*)d_in[7];
  const float* sa_wv = (const float*)d_in[8];
  const float* sa_bv = (const float*)d_in[9];
  const float* sa_wo = (const float*)d_in[10];
  const float* sa_bo = (const float*)d_in[11];
  const float* ln2_g = (const float*)d_in[12];
  const float* ln2_b = (const float*)d_in[13];
  const float* ca_wq = (const float*)d_in[14];
  const float* ca_bq = (const float*)d_in[15];
  const float* ca_wk = (const float*)d_in[16];
  const float* ca_bk = (const float*)d_in[17];
  const float* ca_wv = (const float*)d_in[18];
  const float* ca_bv = (const float*)d_in[19];
  const float* ca_wo = (const float*)d_in[20];
  const float* ca_bo = (const float*)d_in[21];
  const float* ln3_g = (const float*)d_in[22];
  const float* ln3_b = (const float*)d_in[23];
  const float* fc1_w = (const float*)d_in[24];
  const float* fc1_b = (const float*)d_in[25];
  const float* fc2_w = (const float*)d_in[26];
  const float* fc2_b = (const float*)d_in[27];
  float* out = (float*)d_out;

  const int M  = BB * LXD;     // 4096 query rows
  const int Mk = BB * LYD;     // 4096 kv rows
  size_t off = 0;
  auto alloc = [&](size_t bytes) -> void* {
    void* p = (char*)d_ws + off;
    off += (bytes + 255) & ~(size_t)255;
    return p;
  };
  const size_t EE = (size_t)ED * ED;
  unsigned short* w_saq = (unsigned short*)alloc(EE * 2);
  unsigned short* w_sak = (unsigned short*)alloc(EE * 2);
  unsigned short* w_sav = (unsigned short*)alloc(EE * 2);
  unsigned short* w_sao = (unsigned short*)alloc(EE * 2);
  unsigned short* w_caq = (unsigned short*)alloc(EE * 2);
  unsigned short* w_cak = (unsigned short*)alloc(EE * 2);
  unsigned short* w_cav = (unsigned short*)alloc(EE * 2);
  unsigned short* w_cao = (unsigned short*)alloc(EE * 2);
  unsigned short* w_fc1 = (unsigned short*)alloc((size_t)ED * FFD * 2);
  unsigned short* w_fc2 = (unsigned short*)alloc((size_t)FFD * ED * 2);
  unsigned short* y_bf  = (unsigned short*)alloc((size_t)Mk * ED * 2);
  unsigned short* h_bf  = (unsigned short*)alloc((size_t)M * ED * 2);
  float*          tmpf  = (float*)alloc((size_t)M * ED * 4);
  unsigned short* q_bf  = (unsigned short*)alloc((size_t)M * ED * 2);
  unsigned short* k_bf  = (unsigned short*)alloc((size_t)Mk * ED * 2);
  unsigned short* v_bf  = (unsigned short*)alloc((size_t)Mk * ED * 2);
  unsigned short* a_bf  = (unsigned short*)alloc((size_t)M * ED * 2);
  float*          x1    = (float*)alloc((size_t)M * ED * 4);
  float*          x2    = (float*)alloc((size_t)M * ED * 4);
  unsigned short* ff_bf = (unsigned short*)alloc((size_t)M * FFD * 2);

  auto conv = [&](const float* s, unsigned short* d, size_t n) {
    conv_f32_bf16<<<2048, 256, 0, stream>>>(s, d, (int)n);
  };
  conv(sa_wq, w_saq, EE); conv(sa_wk, w_sak, EE); conv(sa_wv, w_sav, EE); conv(sa_wo, w_sao, EE);
  conv(ca_wq, w_caq, EE); conv(ca_wk, w_cak, EE); conv(ca_wv, w_cav, EE); conv(ca_wo, w_cao, EE);
  conv(fc1_w, w_fc1, (size_t)ED * FFD); conv(fc2_w, w_fc2, (size_t)FFD * ED);
  conv(y, y_bf, (size_t)Mk * ED);

  const dim3 gN1024((1024 / 128), (M / 64));
  const dim3 gN4096((4096 / 128), (M / 64));
  const dim3 gAttn(LXD / 64, HH, BB);
  const int ropeBlocks = (M * HH * 32 + 255) / 256;
  const float qscale = 0.125f;  // HD^-0.5

  // ======================= self-attention =======================
  layernorm_bf16<<<M, 256, 0, stream>>>(x, ln1_g, ln1_b, h_bf, ED);
  gemm_bf16_wmma<0><<<gN1024, 256, 0, stream>>>(h_bf, w_saq, sa_bq, nullptr, tmpf, M, ED, ED);
  rope_scale_bf16<<<ropeBlocks, 256, 0, stream>>>(tmpf, q_bf, M, LXD, qscale);
  gemm_bf16_wmma<0><<<gN1024, 256, 0, stream>>>(h_bf, w_sak, sa_bk, nullptr, tmpf, M, ED, ED);
  rope_scale_bf16<<<ropeBlocks, 256, 0, stream>>>(tmpf, k_bf, M, LXD, 1.0f);
  gemm_bf16_wmma<2><<<gN1024, 256, 0, stream>>>(h_bf, w_sav, sa_bv, nullptr, v_bf, M, ED, ED);
  attn_wmma<<<gAttn, 128, 0, stream>>>(q_bf, k_bf, v_bf, a_bf, LXD, LXD, 1);
  gemm_bf16_wmma<1><<<gN1024, 256, 0, stream>>>(a_bf, w_sao, sa_bo, x, x1, M, ED, ED);

  // ======================= cross-attention ======================
  layernorm_bf16<<<M, 256, 0, stream>>>(x1, ln2_g, ln2_b, h_bf, ED);
  gemm_bf16_wmma<0><<<gN1024, 256, 0, stream>>>(h_bf, w_caq, ca_bq, nullptr, tmpf, M, ED, ED);
  rope_scale_bf16<<<ropeBlocks, 256, 0, stream>>>(tmpf, q_bf, M, LXD, qscale);
  gemm_bf16_wmma<0><<<gN1024, 256, 0, stream>>>(y_bf, w_cak, ca_bk, nullptr, tmpf, Mk, ED, ED);
  rope_scale_bf16<<<(Mk * HH * 32 + 255) / 256, 256, 0, stream>>>(tmpf, k_bf, Mk, LYD, 1.0f);
  gemm_bf16_wmma<2><<<gN1024, 256, 0, stream>>>(y_bf, w_cav, ca_bv, nullptr, v_bf, Mk, ED, ED);
  attn_wmma<<<gAttn, 128, 0, stream>>>(q_bf, k_bf, v_bf, a_bf, LXD, LYD, 0);
  gemm_bf16_wmma<1><<<gN1024, 256, 0, stream>>>(a_bf, w_cao, ca_bo, x1, x2, M, ED, ED);

  // ============================ MLP =============================
  layernorm_bf16<<<M, 256, 0, stream>>>(x2, ln3_g, ln3_b, h_bf, ED);
  gemm_bf16_wmma<3><<<gN4096, 256, 0, stream>>>(h_bf, w_fc1, fc1_b, nullptr, ff_bf, M, FFD, ED);
  gemm_bf16_wmma<1><<<gN1024, 256, 0, stream>>>(ff_bf, w_fc2, fc2_b, x2, out, M, ED, FFD);
}
